// Joint_28862180229192
// MI455X (gfx1250) — matrix operand order repaired and checked by reading, not compiled, not run on hardware
//
#include <hip/hip_runtime.h>

typedef __attribute__((ext_vector_type(2)))  float   v2f;
typedef __attribute__((ext_vector_type(8)))  float   v8f;
typedef __attribute__((ext_vector_type(8)))  __bf16  v8bf;
typedef __attribute__((ext_vector_type(16))) __bf16  v16bf;
typedef int v4i __attribute__((__vector_size__(4 * sizeof(int))));

#define VOCAB   1024
#define PRED_H  320
#define ENC_H   1024
#define JOINT_H 640
#define NB      8
#define NT      256
#define NU      65
#define MROWS   (NT * NU)   // 16640 rows per batch in the fused GEMM

#define AS1 __attribute__((address_space(1)))
#define AS3 __attribute__((address_space(3)))

// Async global->LDS staging path (gfx1250): guarded so the kernel still
// compiles (falling back to direct global loads) if the builtin is absent.
#if defined(__has_builtin)
#if __has_builtin(__builtin_amdgcn_global_load_async_to_lds_b128)
#define JOINT_ASYNC 1
#endif
#endif

#if defined(JOINT_ASYNC)
#if __has_builtin(__builtin_amdgcn_s_wait_asynccnt)
#define WAIT_ASYNC(n) __builtin_amdgcn_s_wait_asynccnt(n)
#else
#define WAIT_ASYNC(n) asm volatile("s_wait_asynccnt %0" ::"i"(n) : "memory")
#endif
#endif

// ---------- helpers ----------
__device__ __forceinline__ unsigned short f32_bf16(float x) {
  unsigned int u = __float_as_uint(x);
  unsigned int r = (u + 0x7FFFu + ((u >> 16) & 1u)) >> 16;  // round-nearest-even
  return (unsigned short)r;
}

__device__ __forceinline__ v16bf cat8(v8bf lo, v8bf hi) {
  return __builtin_shufflevector(lo, hi, 0, 1, 2, 3, 4, 5, 6, 7,
                                         8, 9, 10, 11, 12, 13, 14, 15);
}

#if defined(JOINT_ASYNC)
__device__ __forceinline__ void async_cp64B(const unsigned short* gsrc,
                                            unsigned short* ldst) {
  AS1 v4i* gs = (AS1 v4i*)(uintptr_t)gsrc;
  AS3 v4i* ld = (AS3 v4i*)(unsigned int)(uintptr_t)ldst;
  __builtin_amdgcn_global_load_async_to_lds_b128(gs, ld, 0, 0);
  __builtin_amdgcn_global_load_async_to_lds_b128(gs, ld, 16, 0);
  __builtin_amdgcn_global_load_async_to_lds_b128(gs, ld, 32, 0);
  __builtin_amdgcn_global_load_async_to_lds_b128(gs, ld, 48, 0);
}
#endif

// ---------- kernel 1: W2 (640x1024 f32, row-major) -> W2t[n][k] bf16 ----------
__global__ void k_w2bf(const float* __restrict__ W2, unsigned short* __restrict__ W2t) {
  int idx = blockIdx.x * 256 + threadIdx.x;       // idx = n*640 + k
  if (idx >= VOCAB * JOINT_H) return;
  int n = idx / JOINT_H;
  int k = idx - n * JOINT_H;
  W2t[idx] = f32_bf16(W2[(size_t)k * VOCAB + n]);
}

// ---------- kernel 2: hfb = f @ W1[:1024] + b1   (fp32 WMMA 16x16x4) ----------
__global__ void k_hf(const float* __restrict__ f, const float* __restrict__ W1,
                     const float* __restrict__ b1, float* __restrict__ hfb) {
  int tid  = threadIdx.x;
  int wave = tid >> 5, lane = tid & 31;
  int half = lane >> 4, l16 = lane & 15;
  int tile = blockIdx.x * 8 + wave;               // 128 x 40 = 5120 tiles
  int mt = tile / 40;
  int nt = tile - mt * 40;
  int rowA = mt * 16 + l16;
  int colB = nt * 16 + l16;

  v8f c = {0.f, 0.f, 0.f, 0.f, 0.f, 0.f, 0.f, 0.f};
  const float* fp = f  + (size_t)rowA * ENC_H + 2 * half;          // A: k = 2*half + r
  const float* bp = W1 + (size_t)(2 * half) * JOINT_H + colB;      // B: k = 2*half + r, n = l16
#pragma unroll 4
  for (int kk = 0; kk < ENC_H / 4; ++kk) {
    v2f a = *(const v2f*)fp;
    v2f bb;
    bb.x = bp[0];
    bb.y = bp[JOINT_H];
    c = __builtin_amdgcn_wmma_f32_16x16x4_f32(false, a, false, bb, (short)0, c,
                                              false, false);
    fp += 4;
    bp += 4 * JOINT_H;
  }
  float bias = b1[colB];
  float* op = hfb + (size_t)(mt * 16 + 8 * half) * JOINT_H + colB; // C: m = r + 8*half
#pragma unroll
  for (int r = 0; r < 8; ++r) op[(size_t)r * JOINT_H] = c[r] + bias;
}

// ---------- kernel 3: hg = g @ W1[1024:]   (fp32 WMMA, M=520 with guards) ----------
__global__ void k_hg(const float* __restrict__ g, const float* __restrict__ W1,
                     float* __restrict__ hgo) {
  int tid  = threadIdx.x;
  int wave = tid >> 5, lane = tid & 31;
  int half = lane >> 4, l16 = lane & 15;
  int tile = blockIdx.x * 8 + wave;               // 33 x 40 = 1320 tiles
  int mt = tile / 40;
  int nt = tile - mt * 40;
  int rowA = mt * 16 + l16;
  int rA   = rowA < (NB * NU) ? rowA : (NB * NU - 1);
  int colB = nt * 16 + l16;

  v8f c = {0.f, 0.f, 0.f, 0.f, 0.f, 0.f, 0.f, 0.f};
  const float* gp = g  + (size_t)rA * PRED_H + 2 * half;
  const float* bp = W1 + (size_t)(ENC_H + 2 * half) * JOINT_H + colB;
#pragma unroll 4
  for (int kk = 0; kk < PRED_H / 4; ++kk) {
    v2f a = *(const v2f*)gp;
    v2f bb;
    bb.x = bp[0];
    bb.y = bp[JOINT_H];
    c = __builtin_amdgcn_wmma_f32_16x16x4_f32(false, a, false, bb, (short)0, c,
                                              false, false);
    gp += 4;
    bp += 4 * JOINT_H;
  }
  int mbase = mt * 16 + 8 * half;
#pragma unroll
  for (int r = 0; r < 8; ++r) {
    int m = mbase + r;
    if (m < NB * NU) hgo[(size_t)m * JOINT_H + colB] = c[r];
  }
}

// ---------- kernel 4: fused  logits = relu(hfb[t]+hg[u]) @ W2 + b2  (bf16 WMMA) ----
// Workgroup: 64 rows x 1024 vocab. 8 waves = 2(wm) x 4(wn); each wave owns
// 32 rows x 64 cols per N-iteration (2 A frags x 4 B frags -> 8 WMMA/k-step).
__global__ void __launch_bounds__(256) k_joint(
    const float* __restrict__ hfb, const float* __restrict__ hgo,
    const unsigned short* __restrict__ W2t, const float* __restrict__ b2,
    float* __restrict__ out) {
  // 648 = 640 + 8 pad -> row stride 1296 B (16B aligned, 324 dwords = 4 mod 64 banks)
  __shared__ __align__(16) unsigned short hT[64][648];          // 82,944 B
#if defined(JOINT_ASYNC)
  // W2 chunk staging: 256 n-rows x 32 k (64 B) padded to 72 B (18 dwords, conflict-free)
  __shared__ __align__(16) unsigned short BtS[2][256 * 36];     // 36,864 B
#endif

  int tid = threadIdx.x;
  int b   = blockIdx.y;
  int m0  = blockIdx.x * 64;                       // row offset within batch b

  // Build bf16 h-tile: h[r][k] = relu(hfb[b,t,k] + hg[b,u,k])  (b1 folded into hfb)
  for (int idx = tid; idx < 64 * JOINT_H; idx += 256) {
    int r = idx / JOINT_H;
    int k = idx - r * JOINT_H;
    int m = m0 + r;
    int t = m / NU;
    int u = m - t * NU;
    float v = hfb[(size_t)((b << 8) + t) * JOINT_H + k] +
              hgo[(size_t)(b * NU + u) * JOINT_H + k];
    hT[r][k] = f32_bf16(v > 0.f ? v : 0.f);
  }
  __syncthreads();

  int wave = tid >> 5, lane = tid & 31;
  int wm = wave & 1, wn = wave >> 1;
  int half = lane >> 4, l16 = lane & 15;
  const unsigned short* hrow0 = &hT[wm * 32 + l16][0];       // A frag row: m = l16
  const unsigned short* hrow1 = &hT[wm * 32 + 16 + l16][0];

  for (int ni = 0; ni < 4; ++ni) {
    int nbase = ni * 256;
    int nb = nbase + wn * 64;
    v8f zero = {0.f, 0.f, 0.f, 0.f, 0.f, 0.f, 0.f, 0.f};
    v8f acc[2][4];
#pragma unroll
    for (int s = 0; s < 2; ++s)
#pragma unroll
      for (int j = 0; j < 4; ++j) acc[s][j] = zero;

#if defined(JOINT_ASYNC)
    // prologue: stage k-chunk 0 into buffer 0 (thread tid owns n-row nbase+tid)
    async_cp64B(W2t + (size_t)(nbase + tid) * JOINT_H, &BtS[0][tid * 36]);
    for (int kk = 0; kk < JOINT_H / 32; ++kk) {
      int kb = kk * 32;
      int cur = kk & 1;
      if (kk < JOINT_H / 32 - 1) {   // stage next chunk into the other buffer
        async_cp64B(W2t + (size_t)(nbase + tid) * JOINT_H + kb + 32,
                    &BtS[cur ^ 1][tid * 36]);
        WAIT_ASYNC(4);               // current chunk (older 4 ops) complete
      } else {
        WAIT_ASYNC(0);
      }
      __syncthreads();

      v16bf A0 = cat8(*(const v8bf*)(hrow0 + kb + 8 * half),
                      *(const v8bf*)(hrow0 + kb + 16 + 8 * half));
      v16bf A1 = cat8(*(const v8bf*)(hrow1 + kb + 8 * half),
                      *(const v8bf*)(hrow1 + kb + 16 + 8 * half));
#pragma unroll
      for (int j = 0; j < 4; ++j) {
        const unsigned short* bp = &BtS[cur][(wn * 64 + j * 16 + l16) * 36 + 8 * half];
        v16bf Bv = cat8(*(const v8bf*)bp, *(const v8bf*)(bp + 16));
        acc[0][j] = __builtin_amdgcn_wmma_f32_16x16x32_bf16(false, A0, false, Bv,
                                                            (short)0, acc[0][j],
                                                            false, false);
        acc[1][j] = __builtin_amdgcn_wmma_f32_16x16x32_bf16(false, A1, false, Bv,
                                                            (short)0, acc[1][j],
                                                            false, false);
      }
      __syncthreads();               // protect buffer being restaged next iter
    }
#else
    for (int kk = 0; kk < JOINT_H / 32; ++kk) {
      int kb = kk * 32;
      v16bf A0 = cat8(*(const v8bf*)(hrow0 + kb + 8 * half),
                      *(const v8bf*)(hrow0 + kb + 16 + 8 * half));
      v16bf A1 = cat8(*(const v8bf*)(hrow1 + kb + 8 * half),
                      *(const v8bf*)(hrow1 + kb + 16 + 8 * half));
#pragma unroll
      for (int j = 0; j < 4; ++j) {
        const unsigned short* bp =
            W2t + (size_t)(nb + j * 16 + l16) * JOINT_H + kb + 8 * half;
        v16bf Bv = cat8(*(const v8bf*)bp, *(const v8bf*)(bp + 16));
        acc[0][j] = __builtin_amdgcn_wmma_f32_16x16x32_bf16(false, A0, false, Bv,
                                                            (short)0, acc[0][j],
                                                            false, false);
        acc[1][j] = __builtin_amdgcn_wmma_f32_16x16x32_bf16(false, A1, false, Bv,
                                                            (short)0, acc[1][j],
                                                            false, false);
      }
    }
#endif

    // C/D: m = r + 8*half, n = l16 (within 16x16 tile)
#pragma unroll
    for (int s = 0; s < 2; ++s) {
      int mrow = m0 + wm * 32 + s * 16 + 8 * half;
      float* obase = out + ((size_t)b * MROWS + mrow) * VOCAB;
#pragma unroll
      for (int j = 0; j < 4; ++j) {
        int n = nb + j * 16 + l16;
        float bias = b2[n];
#pragma unroll
        for (int r = 0; r < 8; ++r)
          obase[(size_t)r * VOCAB + n] = acc[s][j][r] + bias;
      }
    }
  }
}

// ---------- host launch ----------
extern "C" void kernel_launch(void* const* d_in, const int* in_sizes, int n_in,
                              void* d_out, int out_size, void* d_ws, size_t ws_size,
                              hipStream_t stream) {
  (void)in_sizes; (void)n_in; (void)out_size; (void)ws_size;
  const float* f  = (const float*)d_in[0];
  const float* g  = (const float*)d_in[1];
  const float* W1 = (const float*)d_in[2];
  const float* b1 = (const float*)d_in[3];
  const float* W2 = (const float*)d_in[4];
  const float* b2 = (const float*)d_in[5];
  float* out = (float*)d_out;

  char* ws = (char*)d_ws;
  float*          hfb = (float*)(ws);                       // 2048*640*4 = 5,242,880 B
  float*          hgo = (float*)(ws + 5242880);             //  520*640*4 = 1,331,200 B
  unsigned short* W2t = (unsigned short*)(ws + 6574080);    // 1024*640*2 = 1,310,720 B

  k_w2bf<<<2560, 256, 0, stream>>>(W2, W2t);
  k_hf  <<<640,  256, 0, stream>>>(f, W1, b1, hfb);
  k_hg  <<<165,  256, 0, stream>>>(g, W1, hgo);
  k_joint<<<dim3(260, 8), 256, 0, stream>>>(hfb, hgo, W2t, b2, out);
}
